// DetectorNetwork_16862041604777
// MI455X (gfx1250) — compile-verified, matrix operand order checked by reading, not run on hardware
//
#include <hip/hip_runtime.h>
#include <math.h>

// ---------------------------------------------------------------------------
// Types for CDNA5 WMMA (wave32). bf16 A/B, f32 accumulate.
// ---------------------------------------------------------------------------
typedef __bf16 bf16_t;
typedef __attribute__((ext_vector_type(16))) __bf16 v16bf;
typedef __attribute__((ext_vector_type(8)))  float  v8f;

union FragBF {
  v16bf  v;
  uint4  q[2];
  __bf16 h[16];
};

// ---------------------------------------------------------------------------
// fp32 -> bf16 elementwise convert
// ---------------------------------------------------------------------------
__global__ void k_f32_to_bf16(const void* __restrict__ inp, void* __restrict__ outp, int n) {
  const float* in = (const float*)inp;
  bf16_t* out = (bf16_t*)outp;
  int i = blockIdx.x * blockDim.x + threadIdx.x;
  if (i < n) out[i] = (bf16_t)in[i];
}

// ---------------------------------------------------------------------------
// Pack fp32 weights [O][Ktot] -> bf16 [Opad][Kpad], zero padded.
// Original (O, Cin, kh, kw) flat layout IS the GEMM B^T layout we want.
// ---------------------------------------------------------------------------
__global__ void k_pack_w(const void* __restrict__ inp, void* __restrict__ outp,
                         int O, int Opad, int Ktot, int Kpad) {
  const float* in = (const float*)inp;
  bf16_t* out = (bf16_t*)outp;
  int i = blockIdx.x * blockDim.x + threadIdx.x;
  int tot = Opad * Kpad;
  if (i >= tot) return;
  int o = i / Kpad;
  int k = i - o * Kpad;
  float v = (o < O && k < Ktot) ? in[o * Ktot + k] : 0.0f;
  out[i] = (bf16_t)v;
}

// ---------------------------------------------------------------------------
// im2col: bf16 CHW activations -> bf16 A-matrix [Mpad][Kpad], zero padded.
// Pays the gather/bounds cost ONCE per element so the WMMA loop is pure b128.
// Handles JAX asymmetric 'SAME' via padlo.
// ---------------------------------------------------------------------------
template<int KSZ>
__global__ void k_im2col(const void* __restrict__ inp, void* __restrict__ outp,
                         int Cin, int Hin, int Win, int Hout, int Wout,
                         int stride, int padlo, int Ktot, int Kpad, int Mpad) {
  const bf16_t* in = (const bf16_t*)inp;
  bf16_t* out = (bf16_t*)outp;
  int i = blockIdx.x * blockDim.x + threadIdx.x;
  int tot = Mpad * Kpad;
  if (i >= tot) return;
  int m = i / Kpad;
  int k = i - m * Kpad;
  const int M = Hout * Wout;
  float v = 0.0f;
  if (m < M && k < Ktot) {
    int ci, ky, kx;
    if (KSZ == 1) { ci = k; ky = 0; kx = 0; }
    else {
      ci = k / (KSZ * KSZ);
      int rem = k - ci * (KSZ * KSZ);
      ky = rem / KSZ;
      kx = rem - ky * KSZ;
    }
    int oy = m / Wout;
    int ox = m - oy * Wout;
    int iy = oy * stride + ky - padlo;
    int ix = ox * stride + kx - padlo;
    if (iy >= 0 && iy < Hin && ix >= 0 && ix < Win)
      v = (float)in[(size_t)ci * Hin * Win + (size_t)iy * Win + ix];
  }
  out[i] = (bf16_t)v;
}

// ---------------------------------------------------------------------------
// GEMM conv via v_wmma_f32_16x16x32_bf16, reading the materialized A matrix.
// One wave computes a 32(M) x 64(N) strip: 2 A frags x 4 B frags -> 8 WMMAs
// per k-step with only 12 b128 loads (1.5 loads/WMMA).
// Fragment layout (ISA 7.12.2, 16-bit A 16x32 / B 32x16):
//   lane l (half=l>>4, l16=l&15): row/col = l16,
//   elements 0..7  -> K = kb + 8*half + 0..7      (one aligned 16B load)
//   elements 8..15 -> K = kb + 16 + 8*half + 0..7 (one aligned 16B load)
// D: lane holds column N=l16; VGPR r holds row = 16*tileM + 8*half + r.
// ---------------------------------------------------------------------------
__global__ void __launch_bounds__(32)
k_conv_gemm(const void* __restrict__ Ap, const void* __restrict__ Wp,
            const void* __restrict__ scalep, const void* __restrict__ biasp,
            const void* __restrict__ residp, void* __restrict__ outp,
            int M, int Cout, int Kpad, int relu)
{
  const bf16_t* A     = (const bf16_t*)Ap;
  const bf16_t* W     = (const bf16_t*)Wp;
  const float*  scale = (const float*)scalep;
  const float*  bias  = (const float*)biasp;
  const bf16_t* resid = (const bf16_t*)residp;
  bf16_t*       out   = (bf16_t*)outp;

  const int lane = threadIdx.x;
  const int half = lane >> 4;
  const int l16  = lane & 15;

  const bf16_t* arow0 = A + (size_t)(blockIdx.x * 32 + l16) * Kpad + half * 8;
  const bf16_t* arow1 = arow0 + (size_t)16 * Kpad;
  const bf16_t* brow  = W + (size_t)(blockIdx.y * 64 + l16) * Kpad + half * 8;
  const size_t  bstr  = (size_t)16 * Kpad;   // next N tile

  v8f a00 = {}, a01 = {}, a02 = {}, a03 = {};
  v8f a10 = {}, a11 = {}, a12 = {}, a13 = {};

  for (int kb = 0; kb < Kpad; kb += 32) {
    FragBF fa0, fa1;
    fa0.q[0] = *(const uint4*)(arow0 + kb);
    fa0.q[1] = *(const uint4*)(arow0 + kb + 16);
    fa1.q[0] = *(const uint4*)(arow1 + kb);
    fa1.q[1] = *(const uint4*)(arow1 + kb + 16);

    FragBF fb0, fb1, fb2, fb3;
    fb0.q[0] = *(const uint4*)(brow + kb);
    fb0.q[1] = *(const uint4*)(brow + kb + 16);
    fb1.q[0] = *(const uint4*)(brow + bstr + kb);
    fb1.q[1] = *(const uint4*)(brow + bstr + kb + 16);
    fb2.q[0] = *(const uint4*)(brow + 2 * bstr + kb);
    fb2.q[1] = *(const uint4*)(brow + 2 * bstr + kb + 16);
    fb3.q[0] = *(const uint4*)(brow + 3 * bstr + kb);
    fb3.q[1] = *(const uint4*)(brow + 3 * bstr + kb + 16);

    a00 = __builtin_amdgcn_wmma_f32_16x16x32_bf16(false, fa0.v, false, fb0.v, (short)0, a00, false, false);
    a01 = __builtin_amdgcn_wmma_f32_16x16x32_bf16(false, fa0.v, false, fb1.v, (short)0, a01, false, false);
    a02 = __builtin_amdgcn_wmma_f32_16x16x32_bf16(false, fa0.v, false, fb2.v, (short)0, a02, false, false);
    a03 = __builtin_amdgcn_wmma_f32_16x16x32_bf16(false, fa0.v, false, fb3.v, (short)0, a03, false, false);
    a10 = __builtin_amdgcn_wmma_f32_16x16x32_bf16(false, fa1.v, false, fb0.v, (short)0, a10, false, false);
    a11 = __builtin_amdgcn_wmma_f32_16x16x32_bf16(false, fa1.v, false, fb1.v, (short)0, a11, false, false);
    a12 = __builtin_amdgcn_wmma_f32_16x16x32_bf16(false, fa1.v, false, fb2.v, (short)0, a12, false, false);
    a13 = __builtin_amdgcn_wmma_f32_16x16x32_bf16(false, fa1.v, false, fb3.v, (short)0, a13, false, false);
  }

  // ---- Epilogue: affine + residual + ReLU, bf16 store (CHW output) ----
  v8f accs[2][4] = {{a00, a01, a02, a03}, {a10, a11, a12, a13}};
#pragma unroll
  for (int mt = 0; mt < 2; ++mt) {
#pragma unroll
    for (int j = 0; j < 4; ++j) {
      const int n = blockIdx.y * 64 + j * 16 + l16;   // out channel
      if (n >= Cout) continue;
      float sc = scale ? scale[n] : 1.0f;
      float bi = bias ? bias[n] : 0.0f;
#pragma unroll
      for (int r = 0; r < 8; ++r) {
        int mo = blockIdx.x * 32 + mt * 16 + half * 8 + r;
        if (mo < M) {
          float v = accs[mt][j][r] * sc + bi;
          if (resid) v += (float)resid[(size_t)n * M + mo];
          if (relu && v < 0.0f) v = 0.0f;
          out[(size_t)n * M + mo] = (bf16_t)v;
        }
      }
    }
  }
}

// ---------------------------------------------------------------------------
// 3x3 stride-2 SAME maxpool (pad_lo = 0, pad_hi = 1)
// ---------------------------------------------------------------------------
__global__ void k_maxpool(const void* __restrict__ inp, void* __restrict__ outp,
                          int C, int Hin, int Win, int Hout, int Wout) {
  const bf16_t* in = (const bf16_t*)inp;
  bf16_t* out = (bf16_t*)outp;
  int i = blockIdx.x * blockDim.x + threadIdx.x;
  int tot = C * Hout * Wout;
  if (i >= tot) return;
  int c = i / (Hout * Wout);
  int rem = i - c * (Hout * Wout);
  int oy = rem / Wout;
  int ox = rem - oy * Wout;
  float m = -1e30f;
  for (int dy = 0; dy < 3; ++dy) {
    int iy = oy * 2 + dy;
    if (iy >= Hin) break;
    for (int dx = 0; dx < 3; ++dx) {
      int ix = ox * 2 + dx;
      if (ix >= Win) break;
      float v = (float)in[(size_t)c * Hin * Win + (size_t)iy * Win + ix];
      m = fmaxf(m, v);
    }
  }
  out[i] = (bf16_t)m;
}

// ---------------------------------------------------------------------------
// Anchor decode: sigmoid scores, delta->box, clip, score threshold mask
// ---------------------------------------------------------------------------
#define NANCH 2304   // 16*16*9
__global__ void k_decode(const void* __restrict__ clsp, const void* __restrict__ regp,
                         float* __restrict__ scores, float* __restrict__ boxes) {
  const bf16_t* cls = (const bf16_t*)clsp;
  const bf16_t* reg = (const bf16_t*)regp;
  int a = blockIdx.x * blockDim.x + threadIdx.x;
  if (a >= NANCH) return;
  int ch = a % 9;
  int x  = (a / 9) % 16;
  int y  = a / 144;
  int sp = y * 16 + x;

  float logit = (float)cls[ch * 256 + sp];
  float s = 1.0f / (1.0f + __expf(-logit));

  float d0 = (float)reg[(ch * 4 + 0) * 256 + sp];
  float d1 = (float)reg[(ch * 4 + 1) * 256 + sp];
  float d2 = (float)reg[(ch * 4 + 2) * 256 + sp];
  float d3 = (float)reg[(ch * 4 + 3) * 256 + sp];

  const float SC[3] = {64.0f, 128.0f, 256.0f};
  const float RT[3] = {0.5f, 1.0f, 2.0f};
  float sv = SC[ch / 3], rv = RT[ch % 3];
  float aw = sv / sqrtf(rv);
  float ah = sv * sqrtf(rv);
  float cx = (x + 0.5f) * 32.0f + d0 * aw;
  float cy = (y + 0.5f) * 32.0f + d1 * ah;
  float bw = aw * __expf(fminf(fmaxf(d2, -4.0f), 4.0f));
  float bh = ah * __expf(fminf(fmaxf(d3, -4.0f), 4.0f));

  boxes[a * 4 + 0] = fminf(fmaxf(cx - bw * 0.5f, 0.0f), 512.0f);
  boxes[a * 4 + 1] = fminf(fmaxf(cy - bh * 0.5f, 0.0f), 512.0f);
  boxes[a * 4 + 2] = fminf(fmaxf(cx + bw * 0.5f, 0.0f), 512.0f);
  boxes[a * 4 + 3] = fminf(fmaxf(cy + bh * 0.5f, 0.0f), 512.0f);
  scores[a] = (s > 0.45f) ? s : -1.0f;
}

// ---------------------------------------------------------------------------
// Greedy NMS (score-descending == reference's sorted sequential suppression).
// Single block; everything lives in LDS (~48 KB << 320 KB/WGP).
// ---------------------------------------------------------------------------
__device__ __forceinline__ float iou_pair(float ax1, float ay1, float ax2, float ay2,
                                          float bx1, float by1, float bx2, float by2) {
  float areaA = (ax2 - ax1) * (ay2 - ay1);
  float areaB = (bx2 - bx1) * (by2 - by1);
  float ix1 = fmaxf(ax1, bx1), iy1 = fmaxf(ay1, by1);
  float ix2 = fminf(ax2, bx2), iy2 = fminf(ay2, by2);
  float inter = fmaxf(ix2 - ix1, 0.0f) * fmaxf(iy2 - iy1, 0.0f);
  return inter / (areaA + areaB - inter + 1e-9f);
}

__global__ void __launch_bounds__(256)
k_nms(const float* __restrict__ scores, const float* __restrict__ boxes,
      float* __restrict__ out) {
  __shared__ float ss[NANCH];
  __shared__ float bx[NANCH * 4];
  __shared__ float rv[256];
  __shared__ int   ri[256];
  __shared__ float sBx[4];
  const int tid = threadIdx.x;

  for (int i = tid; i < NANCH; i += 256) {
    ss[i] = scores[i];
    bx[i * 4 + 0] = boxes[i * 4 + 0];
    bx[i * 4 + 1] = boxes[i * 4 + 1];
    bx[i * 4 + 2] = boxes[i * 4 + 2];
    bx[i * 4 + 3] = boxes[i * 4 + 3];
  }
  __syncthreads();

  for (int det = 0; det < 300; ++det) {
    float bm = -1e30f; int bi = 0;
    for (int i = tid; i < NANCH; i += 256) {
      float v = ss[i];
      if (v > bm) { bm = v; bi = i; }
    }
    rv[tid] = bm; ri[tid] = bi;
    __syncthreads();
    for (int s = 128; s > 0; s >>= 1) {
      if (tid < s && rv[tid + s] > rv[tid]) { rv[tid] = rv[tid + s]; ri[tid] = ri[tid + s]; }
      __syncthreads();
    }
    if (tid == 0) {
      int b = ri[0];
      sBx[0] = bx[b * 4 + 0]; sBx[1] = bx[b * 4 + 1];
      sBx[2] = bx[b * 4 + 2]; sBx[3] = bx[b * 4 + 3];
      out[det * 5 + 0] = sBx[0];
      out[det * 5 + 1] = sBx[1];
      out[det * 5 + 2] = sBx[2];
      out[det * 5 + 3] = sBx[3];
      out[det * 5 + 4] = fmaxf(rv[0], 0.0f);
      ss[b] = -1e30f;
    }
    __syncthreads();
    float bscore = rv[0];
    if (bscore > 0.45f) {
      float a0 = sBx[0], a1 = sBx[1], a2 = sBx[2], a3 = sBx[3];
      for (int i = tid; i < NANCH; i += 256) {
        float v = ss[i];
        if (v > 0.45f) {
          if (iou_pair(a0, a1, a2, a3,
                       bx[i * 4 + 0], bx[i * 4 + 1], bx[i * 4 + 2], bx[i * 4 + 3]) > 0.5f)
            ss[i] = -1.0f;
        }
      }
    }
    __syncthreads();
  }
}

// ---------------------------------------------------------------------------
// Host orchestration
// ---------------------------------------------------------------------------
static inline int ceil_div(int a, int b) { return (a + b - 1) / b; }
static inline int roundup32(int x) { return (x + 31) & ~31; }

static void launch_pack(const void* w, void* outw, int O, int Opad, int Ktot, int Kpad,
                        hipStream_t st) {
  int tot = Opad * Kpad;
  k_pack_w<<<ceil_div(tot, 256), 256, 0, st>>>(w, outw, O, Opad, Ktot, Kpad);
}

static void launch_im2col(int ksz, const void* in, void* outA,
                          int Cin, int Hin, int Win, int Hout, int Wout,
                          int stride, int padlo, int Ktot, int Kpad, int Mpad,
                          hipStream_t st) {
  int tot = Mpad * Kpad;
  dim3 grid(ceil_div(tot, 256)), blk(256);
  switch (ksz) {
    case 1: k_im2col<1><<<grid, blk, 0, st>>>(in, outA, Cin, Hin, Win, Hout, Wout, stride, padlo, Ktot, Kpad, Mpad); break;
    case 3: k_im2col<3><<<grid, blk, 0, st>>>(in, outA, Cin, Hin, Win, Hout, Wout, stride, padlo, Ktot, Kpad, Mpad); break;
    case 7: k_im2col<7><<<grid, blk, 0, st>>>(in, outA, Cin, Hin, Win, Hout, Wout, stride, padlo, Ktot, Kpad, Mpad); break;
    default: break;
  }
}

// Full conv: im2col(A) + pack(B) + GEMM with fused epilogue.
static void conv_layer_r(int ksz, const void* in, const void* w, const void* s, const void* b,
                         const void* res, void* out, void* Abuf, void* Wbuf,
                         int Cin, int Cout, int Opad,
                         int Hin, int Win, int Hout, int Wout,
                         int stride, int padlo, int relu, hipStream_t st) {
  int Ktot = Cin * ksz * ksz;
  int Kpad = roundup32(Ktot);
  int M    = Hout * Wout;
  int Mpad = (M + 31) & ~31;                 // M tiles of 32 (all layers: exact)
  launch_im2col(ksz, in, Abuf, Cin, Hin, Win, Hout, Wout, stride, padlo, Ktot, Kpad, Mpad, st);
  launch_pack(w, Wbuf, Cout, Opad, Ktot, Kpad, st);
  dim3 grid(Mpad / 32, Opad / 64), blk(32);
  k_conv_gemm<<<grid, blk, 0, st>>>(Abuf, Wbuf, s, b, res, out, M, Cout, Kpad, relu);
}

extern "C" void kernel_launch(void* const* d_in, const int* in_sizes, int n_in,
                              void* d_out, int out_size, void* d_ws, size_t ws_size,
                              hipStream_t stream) {
  (void)in_sizes; (void)n_in; (void)out_size; (void)ws_size;

  // ---- workspace carve-out (~45 MB) ----
  char* ws = (char*)d_ws;
  size_t off = 0;
  auto alloc = [&](size_t bytes) -> void* {
    void* p = ws + off;
    off += (bytes + 255) & ~(size_t)255;
    return p;
  };
  void* xbf    = alloc((size_t)3 * 512 * 512 * 2);
  void* stemO  = alloc((size_t)64 * 256 * 256 * 2);
  void* bufA   = alloc((size_t)64 * 128 * 128 * 2);
  void* bufB   = alloc((size_t)64 * 128 * 128 * 2);
  void* bufC   = alloc((size_t)64 * 128 * 128 * 2);
  void* bufD   = alloc((size_t)64 * 128 * 128 * 2);
  void* wpack  = alloc((size_t)512 * 4608 * 2);     // biggest packed weights
  void* imbuf  = alloc((size_t)65536 * 160 * 2);    // biggest im2col (stem), 21 MB
  void* clsB   = alloc((size_t)64 * 256 * 2);
  void* regB   = alloc((size_t)64 * 256 * 2);
  float* scBuf = (float*)alloc((size_t)NANCH * 4);
  float* bxBuf = (float*)alloc((size_t)NANCH * 4 * 4);

  // ---- input unpacking (setup_inputs dict insertion order) ----
  int idx = 0;
  const void* X      = d_in[idx++];
  const void* stem_w = d_in[idx++];
  const void* stem_s = d_in[idx++];
  const void* stem_b = d_in[idx++];

  // x -> bf16
  k_f32_to_bf16<<<ceil_div(3 * 512 * 512, 256), 256, 0, stream>>>(X, xbf, 3 * 512 * 512);

  // stem: 7x7 s2 pad_lo=2 (SAME on 512: total pad 5), 3->64, 512->256, affine+ReLU
  conv_layer_r(7, xbf, stem_w, stem_s, stem_b, nullptr, stemO, imbuf, wpack,
               3, 64, 64, 512, 512, 256, 256, 2, 2, 1, stream);

  // maxpool 3x3 s2 SAME -> 64x128x128
  k_maxpool<<<ceil_div(64 * 128 * 128, 256), 256, 0, stream>>>(stemO, bufA, 64, 256, 256, 128, 128);

  void* cur = bufA;
  void* nxt = bufD;
  int Hc = 128, Cinc = 64;
  const int CFGc[4] = {64, 128, 256, 512};
  const int CFGn[4] = {3, 4, 6, 3};
  const int CFGs[4] = {1, 2, 2, 2};

  for (int si = 0; si < 4; ++si) {
    int c = CFGc[si], nblk = CFGn[si], st0 = CFGs[si];
    for (int bi = 0; bi < nblk; ++bi) {
      int stride = (bi == 0) ? st0 : 1;
      const void* w1 = d_in[idx++]; const void* s1 = d_in[idx++]; const void* b1 = d_in[idx++];
      const void* w2 = d_in[idx++]; const void* s2 = d_in[idx++]; const void* b2 = d_in[idx++];
      const void* wpw = nullptr; const void* sp = nullptr; const void* bp = nullptr;
      bool haswp = (bi == 0) && (stride != 1 || Cinc != c);
      if (haswp) { wpw = d_in[idx++]; sp = d_in[idx++]; bp = d_in[idx++]; }
      int Hout = Hc / stride;

      // conv1: 3x3, affine + ReLU (SAME: pad_lo = 1 for s1, 0 for s2)
      conv_layer_r(3, cur, w1, s1, b1, nullptr, bufB, imbuf, wpack,
                   Cinc, c, c, Hc, Hc, Hout, Hout, stride,
                   (stride == 1) ? 1 : 0, 1, stream);
      // shortcut (1x1 stride-2 SAME -> pad 0)
      const void* res = cur;
      if (haswp) {
        conv_layer_r(1, cur, wpw, sp, bp, nullptr, bufC, imbuf, wpack,
                     Cinc, c, c, Hc, Hc, Hout, Hout, stride, 0, 0, stream);
        res = bufC;
      }
      // conv2: 3x3 s1, affine + residual + ReLU
      conv_layer_r(3, bufB, w2, s2, b2, res, nxt, imbuf, wpack,
                   c, c, c, Hout, Hout, Hout, Hout, 1, 1, 1, stream);

      void* t = cur; cur = nxt; nxt = t;
      Hc = Hout; Cinc = c;
    }
  }

  const void* rpn_w = d_in[idx++]; const void* rpn_b = d_in[idx++];
  const void* cls_w = d_in[idx++]; const void* cls_b = d_in[idx++];
  const void* reg_w = d_in[idx++]; const void* reg_b = d_in[idx++];

  // RPN 3x3 512->512 on 16x16, bias + ReLU (scale = null -> 1.0)
  conv_layer_r(3, cur, rpn_w, nullptr, rpn_b, nullptr, bufB, imbuf, wpack,
               512, 512, 512, 16, 16, 16, 16, 1, 1, 1, stream);
  // cls 1x1 512->9 (packed to 64 out rows, store-guarded to 9)
  conv_layer_r(1, bufB, cls_w, nullptr, cls_b, nullptr, clsB, imbuf, wpack,
               512, 9, 64, 16, 16, 16, 16, 1, 0, 0, stream);
  // reg 1x1 512->36 (packed to 64 out rows, store-guarded to 36)
  conv_layer_r(1, bufB, reg_w, nullptr, reg_b, nullptr, regB, imbuf, wpack,
               512, 36, 64, 16, 16, 16, 16, 1, 0, 0, stream);

  k_decode<<<ceil_div(NANCH, 256), 256, 0, stream>>>(clsB, regB, scBuf, bxBuf);
  k_nms<<<1, 256, 0, stream>>>(scBuf, bxBuf, (float*)d_out);
}
